// BlockWrapper_67035849556271
// MI455X (gfx1250) — compile-verified, matrix operand order checked by reading, not run on hardware
//
#include <hip/hip_runtime.h>

// ---------------- dims ----------------
#define T_DIM   256
#define B_DIM   64
#define NHID    1536
#define NB      6
#define BS      256
#define ATT_OUT 1024
#define DK      64
#define NHC     4
#define DKC     16
#define MROWS   16384   // T*B

typedef __bf16 v16bf __attribute__((ext_vector_type(16)));
typedef float  v8f   __attribute__((ext_vector_type(8)));

union Frag { v16bf v; uint4 u[2]; __bf16 h[16]; };

static __device__ __forceinline__ v8f wmma_bf16(v16bf a, v16bf b, v8f c) {
    return __builtin_amdgcn_wmma_f32_16x16x32_bf16(false, a, false, b, (short)0, c, false, false);
}

// ---------------------------------------------------------------------------
// Pack fp32 (K x N) matrices into bf16 WMMA B-fragment order:
//   per mat: [ntile][kstep][lane(32)][16 bf16]  (contiguous 32B per lane)
//   value j<8 : W[kstep*32 + 8*(lane>>4) + j       ][ntile*16 + (lane&15)]
//   value j>=8: W[kstep*32 + 16 + 8*(lane>>4) + j-8][ntile*16 + (lane&15)]
// ---------------------------------------------------------------------------
__global__ void k_pack(const float* __restrict__ src, __bf16* __restrict__ dst,
                       int K, int N, int nmats) {
    int ntiles = N >> 4, ksteps = K >> 5;
    long total = (long)nmats * ntiles * ksteps * 32;
    long id = (long)blockIdx.x * blockDim.x + threadIdx.x;
    if (id >= total) return;
    int lane = (int)(id & 31);
    long t = id >> 5;
    int ks = (int)(t % ksteps); t /= ksteps;
    int nt = (int)(t % ntiles); int mat = (int)(t / ntiles);
    const float* S = src + (size_t)mat * K * N;
    __bf16* D = dst + ((((size_t)mat * ntiles + nt) * ksteps + ks) * 32 + lane) * 16;
    int col = nt * 16 + (lane & 15);
    int row = ks * 32 + 8 * (lane >> 4);
#pragma unroll
    for (int j = 0; j < 8; ++j) D[j]     = (__bf16)S[(size_t)(row + j) * N + col];
#pragma unroll
    for (int j = 0; j < 8; ++j) D[8 + j] = (__bf16)S[(size_t)(row + 16 + j) * N + col];
}

// h -> h_cur (fp32) and h_bf16
__global__ void k_init(const float* __restrict__ h, float* __restrict__ h_cur,
                       __bf16* __restrict__ h_bf) {
    int i = blockIdx.x * 256 + threadIdx.x;   // 98304 total
    float v = h[i];
    h_cur[i] = v;
    h_bf[i] = (__bf16)v;
}

// gic[g][k][h] = bv @ wi[g,k] + bi[g][k*BS+h]
__global__ void k_gic(const float* __restrict__ bv, const float* __restrict__ wi,
                      const float* __restrict__ bi, float* __restrict__ gic) {
    int id = blockIdx.x * 256 + threadIdx.x;  // 4608 total
    int g = id / NHID, r = id % NHID;
    int kb = r / BS, hc = r % BS;
    float acc = bi[g * NHID + r];
    const float* W = wi + ((size_t)(g * NB + kb) * ATT_OUT) * BS + hc;
    for (int i = 0; i < ATT_OUT; ++i) acc += bv[i] * W[(size_t)i * BS];
    gic[id] = acc;
}

// ---------------------------------------------------------------------------
// Generic WMMA GEMM: C[M x N] = A[M x K] @ B (packed bf16) (+bias per col)
// grid = (N/16, M/128, nmats), block = 256 (8 waves, 1 mtile each)
// ---------------------------------------------------------------------------
template<bool ABF16, bool OBF16>
__global__ __launch_bounds__(256)
void k_gemm(const void* __restrict__ Av, int lda, const __bf16* __restrict__ Bp,
            void* __restrict__ Cv, int ldc, int K, const float* __restrict__ bias,
            long aStride, long bStride, long cStride) {
    int wave = threadIdx.x >> 5, lane = threadIdx.x & 31;
    int half = lane >> 4, ln = lane & 15;
    int ksteps = K >> 5;
    size_t m0 = ((size_t)blockIdx.y * 8 + wave) * 16;
    int mat = blockIdx.z;
    const __bf16* B = Bp + (size_t)mat * bStride + (size_t)blockIdx.x * ksteps * 512;
    size_t row = m0 + ln;
    v8f acc = {};
    if (ABF16) {
        const __bf16* A = (const __bf16*)Av + (size_t)mat * aStride + row * lda + 8 * half;
        for (int ks = 0; ks < ksteps; ++ks) {
            Frag a, b;
            const __bf16* ap = A + ks * 32;
            a.u[0] = *(const uint4*)ap;
            a.u[1] = *(const uint4*)(ap + 16);
            const __bf16* bp = B + (size_t)ks * 512 + lane * 16;
            b.u[0] = *(const uint4*)bp;
            b.u[1] = *(const uint4*)(bp + 8);
            acc = wmma_bf16(a.v, b.v, acc);
        }
    } else {
        const float* A = (const float*)Av + (size_t)mat * aStride + row * lda + 8 * half;
        for (int ks = 0; ks < ksteps; ++ks) {
            Frag a, b;
            const float* ap = A + ks * 32;
#pragma unroll
            for (int j = 0; j < 8; ++j) a.h[j] = (__bf16)ap[j];
#pragma unroll
            for (int j = 0; j < 8; ++j) a.h[8 + j] = (__bf16)ap[16 + j];
            const __bf16* bp = B + (size_t)ks * 512 + lane * 16;
            b.u[0] = *(const uint4*)bp;
            b.u[1] = *(const uint4*)(bp + 8);
            acc = wmma_bf16(a.v, b.v, acc);
        }
    }
    int col = blockIdx.x * 16 + ln;
    float bb = bias ? bias[col] : 0.f;
#pragma unroll
    for (int j = 0; j < 8; ++j) {
        size_t m = m0 + j + 8 * half;
        float v = acc[j] + bb;
        if (OBF16) ((__bf16*)Cv + (size_t)mat * cStride)[m * ldc + col] = (__bf16)v;
        else       ((float*)Cv + (size_t)mat * cStride)[m * ldc + col] = v;
    }
}

// ---------------------------------------------------------------------------
// Fused per-timestep kernel: grid = 4 WGs (16 batch rows each), block = 1024
// ---------------------------------------------------------------------------
__global__ __launch_bounds__(1024)
void k_step(int t,
            float* __restrict__ h_cur, __bf16* __restrict__ h_bf, float* __restrict__ hx_ws,
            const float* __restrict__ kx, const float* __restrict__ gic,
            const __bf16* __restrict__ p_wh, const __bf16* __restrict__ p_wq,
            const __bf16* __restrict__ p_cq, const __bf16* __restrict__ p_ck,
            const __bf16* __restrict__ p_cv, const __bf16* __restrict__ p_cfc,
            const float* __restrict__ bq, const float* __restrict__ bk,
            const float* __restrict__ bh,
            const float* __restrict__ cbq, const float* __restrict__ cbk,
            const float* __restrict__ cbv, const float* __restrict__ cbfc,
            const float* __restrict__ ln_g, const float* __restrict__ ln_b,
            const float* __restrict__ GIX, float* __restrict__ out) {
    __shared__ __bf16 s_hb[16 * NHID];    // h (bf16) for 16 batch rows
    __shared__ __bf16 s_hg[16 * NHID];    // r-gate, then hg (=hx) bf16
    __shared__ __bf16 s_mix[16 * NHID];   // q(f32) / z-gate / q2,k2,v2 / o2
    __shared__ float s_att1[96], s_att0[96], s_mask[96], s_sum[96], s_sumsq[96];

    const int tid = threadIdx.x;
    const int wave = tid >> 5, lane = tid & 31;
    const int half = lane >> 4, ln = lane & 15;
    const int b0 = blockIdx.x * 16;

    // ---- phase 0: load h (bf16) to LDS, zero LN stats, prefetch next GIX ----
    {
        const uint* src = (const uint*)(h_bf + (size_t)b0 * NHID);
        uint* dst = (uint*)s_hb;
        for (int i = tid; i < 16 * NHID / 2; i += 1024) dst[i] = src[i];
        if (tid < 96) { s_sum[tid] = 0.f; s_sumsq[tid] = 0.f; }
        if (t + 1 < T_DIM && tid < 288) {
            int mat = tid / 16, bl = tid % 16;
            __builtin_prefetch(GIX + ((size_t)mat * MROWS + (size_t)(t + 1) * B_DIM + b0 + bl) * BS, 0, 1);
        }
    }
    __syncthreads();

    // ---- phase 1: q = hb@wq + bq via WMMA (rows rr = bl*6+kb, 96x64) ----
    float* s_q = (float*)s_mix;                      // 96x64 f32
    {
        int task = wave;                             // 24 tasks: mt(6) x nt(4)
        if (task < 24) {
            int mt = task / 4, nt = task % 4;
            int rr = mt * 16 + ln, bl = rr / 6, kb = rr - 6 * bl;
            const __bf16* abase0 = s_hb + (size_t)bl * NHID + kb * BS + 8 * half;
            v8f acc = {};
            for (int ks = 0; ks < 8; ++ks) {
                Frag a, b;
                const __bf16* ap = abase0 + ks * 32;
                a.u[0] = *(const uint4*)ap; a.u[1] = *(const uint4*)(ap + 16);
                const __bf16* bp = p_wq + ((size_t)(nt * 8 + ks) * 32 + lane) * 16;
                b.u[0] = *(const uint4*)bp; b.u[1] = *(const uint4*)(bp + 8);
                acc = wmma_bf16(a.v, b.v, acc);
            }
            int col = nt * 16 + ln;
            float bb = bq[col];
#pragma unroll
            for (int j = 0; j < 8; ++j) {
                int r2 = mt * 16 + j + 8 * half;
                s_q[r2 * 64 + col] = acc[j] + bb;
            }
        }
    }
    __syncthreads();
    // input attention over [null, x]: softmax over 2 keys (temperature 8)
    if (tid < 96) {
        int bl = tid / NB;
        const float* qp = s_q + tid * DK;
        const float* kxp = kx + ((size_t)t * B_DIM + b0 + bl) * DK;
        float l0 = 0.f, l1 = 0.f;
        for (int d = 0; d < DK; ++d) { l0 += qp[d] * bk[d]; l1 += qp[d] * kxp[d]; }
        l0 *= 0.125f; l1 *= 0.125f;
        float a1 = 1.f / (1.f + __expf(l0 - l1));
        s_att1[tid] = a1; s_att0[tid] = 1.f - a1;
    }
    __syncthreads();
    if (tid < 16) {   // 2 largest att0 -> inactive (ties: lowest index)
        int i1 = 0; float v1 = -1e30f;
        for (int k = 0; k < NB; ++k) { float v = s_att0[tid * NB + k]; if (v > v1) { v1 = v; i1 = k; } }
        int i2 = 0; float v2 = -1e30f;
        for (int k = 0; k < NB; ++k) if (k != i1) { float v = s_att0[tid * NB + k]; if (v > v2) { v2 = v; i2 = k; } }
        for (int k = 0; k < NB; ++k) s_mask[tid * NB + k] = (k == i1 || k == i2) ? 0.f : 1.f;
    }
    __syncthreads();

    // ---- phase 2a: r and z gates (gh = hb@wh + bh ; gi = gic + att1*GIX) ----
    for (int it = 0; it < 6; ++it) {
        int task = wave + 32 * it;                   // 192 tasks: g(2) x kb(6) x nt(16)
        int g = task / 96, rem = task % 96;
        int kb = rem / 16, nt = rem % 16;
        v8f acc = {};
        const __bf16* abase0 = s_hb + (size_t)ln * NHID + kb * BS + 8 * half;
        const __bf16* Bm = p_wh + (size_t)(g * NB + kb) * (BS * BS);
        for (int ks = 0; ks < 8; ++ks) {
            Frag a, b;
            const __bf16* ap = abase0 + ks * 32;
            a.u[0] = *(const uint4*)ap; a.u[1] = *(const uint4*)(ap + 16);
            const __bf16* bp = Bm + ((size_t)(nt * 8 + ks) * 32 + lane) * 16;
            b.u[0] = *(const uint4*)bp; b.u[1] = *(const uint4*)(bp + 8);
            acc = wmma_bf16(a.v, b.v, acc);
        }
        int col = nt * 16 + ln, idx = kb * BS + col;
#pragma unroll
        for (int j = 0; j < 8; ++j) {
            int bl = j + 8 * half;
            float gh = acc[j] + bh[g * NHID + idx];
            float gi = gic[(g * NB + kb) * BS + col] +
                       s_att1[bl * NB + kb] *
                       GIX[((size_t)(g * NB + kb) * MROWS + (size_t)t * B_DIM + b0 + bl) * BS + col];
            float v = 1.f / (1.f + __expf(-(gi + gh)));
            if (g == 0) s_hg[bl * NHID + idx] = (__bf16)v;     // r
            else        s_mix[bl * NHID + idx] = (__bf16)v;    // z
        }
    }
    __syncthreads();

    // ---- phase 2b: n gate + GRU combine -> hx ; s_hg := hx (bf16) ----
    for (int it = 0; it < 3; ++it) {
        int task = wave + 32 * it;                   // 96 tasks: kb x nt
        int kb = task / 16, nt = task % 16;
        v8f acc = {};
        const __bf16* abase0 = s_hb + (size_t)ln * NHID + kb * BS + 8 * half;
        const __bf16* Bm = p_wh + (size_t)(2 * NB + kb) * (BS * BS);
        for (int ks = 0; ks < 8; ++ks) {
            Frag a, b;
            const __bf16* ap = abase0 + ks * 32;
            a.u[0] = *(const uint4*)ap; a.u[1] = *(const uint4*)(ap + 16);
            const __bf16* bp = Bm + ((size_t)(nt * 8 + ks) * 32 + lane) * 16;
            b.u[0] = *(const uint4*)bp; b.u[1] = *(const uint4*)(bp + 8);
            acc = wmma_bf16(a.v, b.v, acc);
        }
        int col = nt * 16 + ln, idx = kb * BS + col;
#pragma unroll
        for (int j = 0; j < 8; ++j) {
            int bl = j + 8 * half;
            float ghn = acc[j] + bh[2 * NHID + idx];
            float gin = gic[(2 * NB + kb) * BS + col] +
                        s_att1[bl * NB + kb] *
                        GIX[((size_t)(2 * NB + kb) * MROWS + (size_t)t * B_DIM + b0 + bl) * BS + col];
            float rv = (float)s_hg[bl * NHID + idx];
            float zv = (float)s_mix[bl * NHID + idx];
            float nv = tanhf(gin + rv * ghn);
            float hbv = h_cur[(size_t)(b0 + bl) * NHID + idx];
            float hx = (1.f - zv) * nv + zv * hbv;
            hx_ws[(size_t)(b0 + bl) * NHID + idx] = hx;
            s_hg[bl * NHID + idx] = (__bf16)hx;      // hg == hx (blocked_grad fwd identity)
        }
    }
    __syncthreads();

    // ---- phase 3: comm q2/k2/v2 = hg@{cq,ck,cv} + bias ----
    float* s_q2 = (float*)s_mix;                     // 96x64 f32
    __bf16* s_k2 = s_mix + 12288;                    // 96x64 bf16
    __bf16* s_v2 = s_mix + 18432;                    // 96x64 bf16
    for (int it = 0; it < 3; ++it) {
        int task = wave + 32 * it;                   // 72 tasks: mat(3) x mt(6) x nt(4)
        if (task < 72) {
            int mat = task / 24, rem = task % 24;
            int mt = rem / 4, nt = rem % 4;
            const __bf16* Bm = (mat == 0) ? p_cq : (mat == 1) ? p_ck : p_cv;
            const float* bias = (mat == 0) ? cbq : (mat == 1) ? cbk : cbv;
            int rr = mt * 16 + ln, bl = rr / 6, kb = rr - 6 * bl;
            const __bf16* abase0 = s_hg + (size_t)bl * NHID + kb * BS + 8 * half;
            v8f acc = {};
            for (int ks = 0; ks < 8; ++ks) {
                Frag a, b;
                const __bf16* ap = abase0 + ks * 32;
                a.u[0] = *(const uint4*)ap; a.u[1] = *(const uint4*)(ap + 16);
                const __bf16* bp = Bm + ((size_t)(nt * 8 + ks) * 32 + lane) * 16;
                b.u[0] = *(const uint4*)bp; b.u[1] = *(const uint4*)(bp + 8);
                acc = wmma_bf16(a.v, b.v, acc);
            }
            int col = nt * 16 + ln;
#pragma unroll
            for (int j = 0; j < 8; ++j) {
                int r2 = mt * 16 + j + 8 * half;
                float v = acc[j] + bias[col];
                if (mat == 0)      s_q2[r2 * 64 + col] = v;
                else if (mat == 1) s_k2[r2 * 64 + col] = (__bf16)v;
                else               s_v2[r2 * 64 + col] = (__bf16)v;
            }
        }
    }
    __syncthreads();

    // ---- phase 4: 4-head attention over NB=6 blocks (temperature 4) ----
    float oacc[DKC];
    int bl4 = 0, hh4 = 0, qi4 = 0;
    if (tid < 384) {
        bl4 = tid / 24; int rem = tid % 24; hh4 = rem / 6; qi4 = rem % 6;
        float lg[NB], mx = -1e30f;
        const float* qrow = s_q2 + (bl4 * NB + qi4) * 64 + hh4 * DKC;
        for (int kk = 0; kk < NB; ++kk) {
            float s = 0.f;
            const __bf16* krow = s_k2 + (bl4 * NB + kk) * 64 + hh4 * DKC;
            for (int d = 0; d < DKC; ++d) s += qrow[d] * (float)krow[d];
            lg[kk] = s * 0.25f;
            mx = fmaxf(mx, lg[kk]);
        }
        float den = 0.f;
        for (int kk = 0; kk < NB; ++kk) { lg[kk] = __expf(lg[kk] - mx); den += lg[kk]; }
        float inv = 1.f / den;
        for (int d = 0; d < DKC; ++d) oacc[d] = 0.f;
        for (int kk = 0; kk < NB; ++kk) {
            float a = lg[kk] * inv;
            const __bf16* vrow = s_v2 + (bl4 * NB + kk) * 64 + hh4 * DKC;
            for (int d = 0; d < DKC; ++d) oacc[d] += a * (float)vrow[d];
        }
    }
    __syncthreads();
    __bf16* s_o2 = s_mix;                            // 96x64 bf16 (overwrites q2)
    if (tid < 384)
        for (int d = 0; d < DKC; ++d) s_o2[(bl4 * NB + qi4) * 64 + hh4 * DKC + d] = (__bf16)oacc[d];
    __syncthreads();

    // ---- phase 5: res = o2@cfc + cbfc + hx ; LayerNorm ; gate ; write out ----
    float resv[24];
    for (int it = 0; it < 3; ++it) {
        int task = wave + 32 * it;                   // 96 tasks: mt(6) x nt(16)
        int mt = task / 16, nt = task % 16;
        int arow = mt * 16 + ln;
        const __bf16* abase0 = s_o2 + (size_t)arow * 64 + 8 * half;
        v8f acc = {};
        for (int ks = 0; ks < 2; ++ks) {
            Frag a, b;
            const __bf16* ap = abase0 + ks * 32;
            a.u[0] = *(const uint4*)ap; a.u[1] = *(const uint4*)(ap + 16);
            const __bf16* bp = p_cfc + ((size_t)(nt * 2 + ks) * 32 + lane) * 16;
            b.u[0] = *(const uint4*)bp; b.u[1] = *(const uint4*)(bp + 8);
            acc = wmma_bf16(a.v, b.v, acc);
        }
        int col = nt * 16 + ln;
#pragma unroll
        for (int j = 0; j < 8; ++j) {
            int rr = mt * 16 + j + 8 * half;
            int bl = rr / 6, kb = rr - 6 * bl, idx = kb * BS + col;
            float res = acc[j] + cbfc[col] + hx_ws[(size_t)(b0 + bl) * NHID + idx];
            resv[it * 8 + j] = res;
            float s1 = res, s2 = res * res;
#pragma unroll
            for (int off = 1; off < 16; off <<= 1) {
                s1 += __shfl_xor(s1, off, 16);
                s2 += __shfl_xor(s2, off, 16);
            }
            if (ln == 0) { atomicAdd(&s_sum[rr], s1); atomicAdd(&s_sumsq[rr], s2); }
        }
    }
    __syncthreads();
    for (int it = 0; it < 3; ++it) {
        int task = wave + 32 * it;
        int mt = task / 16, nt = task % 16;
        int col = nt * 16 + ln;
        float lg = ln_g[col], lb = ln_b[col];
#pragma unroll
        for (int j = 0; j < 8; ++j) {
            int rr = mt * 16 + j + 8 * half;
            int bl = rr / 6, kb = rr - 6 * bl, idx = kb * BS + col;
            float mu = s_sum[rr] * (1.f / 256.f);
            float var = s_sumsq[rr] * (1.f / 256.f) - mu * mu;
            float rstd = rsqrtf(var + 1e-5f);
            float res = resv[it * 8 + j];
            float hx = hx_ws[(size_t)(b0 + bl) * NHID + idx];
            float hxf = hx + (res - mu) * rstd * lg + lb;
            float m = s_mask[bl * NB + kb];
            float hbv = h_cur[(size_t)(b0 + bl) * NHID + idx];
            float hn = m * hxf + (1.f - m) * hbv;
            out[((size_t)t * B_DIM + b0 + bl) * NHID + idx] = hn;
            h_cur[(size_t)(b0 + bl) * NHID + idx] = hn;
            h_bf[(size_t)(b0 + bl) * NHID + idx] = (__bf16)hn;
        }
    }
}

// ---------------------------------------------------------------------------
// host side
// ---------------------------------------------------------------------------
extern "C" void kernel_launch(void* const* d_in, const int* in_sizes, int n_in,
                              void* d_out, int out_size, void* d_ws, size_t ws_size,
                              hipStream_t stream) {
    const float* inp  = (const float*)d_in[0];
    const float* h    = (const float*)d_in[1];
    const float* wq   = (const float*)d_in[2];
    const float* bq   = (const float*)d_in[3];
    const float* wk   = (const float*)d_in[4];
    const float* bk   = (const float*)d_in[5];
    const float* wv   = (const float*)d_in[6];
    const float* bv   = (const float*)d_in[7];
    const float* wi   = (const float*)d_in[8];
    const float* wh   = (const float*)d_in[9];
    const float* bi   = (const float*)d_in[10];
    const float* bh   = (const float*)d_in[11];
    const float* cq   = (const float*)d_in[12];
    const float* cbq  = (const float*)d_in[13];
    const float* ck   = (const float*)d_in[14];
    const float* cbk  = (const float*)d_in[15];
    const float* cv   = (const float*)d_in[16];
    const float* cbv  = (const float*)d_in[17];
    const float* cfc  = (const float*)d_in[18];
    const float* cbfc = (const float*)d_in[19];
    const float* ln_g = (const float*)d_in[20];
    const float* ln_b = (const float*)d_in[21];
    float* out = (float*)d_out;

    // workspace carve-out (~350 MB required)
    char* ws = (char*)d_ws;
    size_t off = 0;
    auto alloc = [&](size_t bytes) -> char* {
        char* p = ws + off;
        off += (bytes + 255) & ~(size_t)255;
        return p;
    };
    float*  h_cur  = (float*)alloc((size_t)B_DIM * NHID * 4);
    __bf16* h_bf   = (__bf16*)alloc((size_t)B_DIM * NHID * 2);
    float*  hx_ws  = (float*)alloc((size_t)B_DIM * NHID * 4);
    float*  kx     = (float*)alloc((size_t)MROWS * DK * 4);
    float*  gic    = (float*)alloc((size_t)3 * NB * BS * 4);
    __bf16* p_wk   = (__bf16*)alloc((size_t)NHID * DK * 2);
    __bf16* p_wv   = (__bf16*)alloc((size_t)NHID * ATT_OUT * 2);
    __bf16* p_wi   = (__bf16*)alloc((size_t)3 * NB * ATT_OUT * BS * 2);
    __bf16* p_wh   = (__bf16*)alloc((size_t)3 * NB * BS * BS * 2);
    __bf16* p_wq   = (__bf16*)alloc((size_t)BS * DK * 2);
    __bf16* p_cq   = (__bf16*)alloc((size_t)BS * 64 * 2);
    __bf16* p_ck   = (__bf16*)alloc((size_t)BS * 64 * 2);
    __bf16* p_cv   = (__bf16*)alloc((size_t)BS * 64 * 2);
    __bf16* p_cfc  = (__bf16*)alloc((size_t)64 * BS * 2);
    __bf16* vx     = (__bf16*)alloc((size_t)MROWS * ATT_OUT * 2);
    float*  GIX    = (float*)alloc((size_t)3 * NB * MROWS * BS * 4);
    (void)ws_size; (void)in_sizes; (void)n_in; (void)out_size;

    // init h state
    k_init<<<(B_DIM * NHID) / 256, 256, 0, stream>>>(h, h_cur, h_bf);

    // pack weights into WMMA B-fragment layout (bf16)
    auto pack = [&](const float* s, __bf16* d, int K, int N, int nm) {
        long total = (long)nm * (N / 16) * (K / 32) * 32;
        k_pack<<<(unsigned)((total + 255) / 256), 256, 0, stream>>>(s, d, K, N, nm);
    };
    pack(wk, p_wk, NHID, DK, 1);
    pack(wv, p_wv, NHID, ATT_OUT, 1);
    pack(wi, p_wi, ATT_OUT, BS, 3 * NB);
    pack(wh, p_wh, BS, BS, 3 * NB);
    pack(wq, p_wq, BS, DK, 1);
    pack(cq, p_cq, BS, 64, 1);
    pack(ck, p_ck, BS, 64, 1);
    pack(cv, p_cv, BS, 64, 1);
    pack(cfc, p_cfc, 64, BS, 1);

    // gic = bv@wi + bi
    k_gic<<<(3 * NB * BS) / 256, 256, 0, stream>>>(bv, wi, bi, gic);

    // kx = x@wk + bk    (M=16384, K=1536, N=64, fp32 out)
    k_gemm<false, false><<<dim3(DK / 16, MROWS / 128, 1), 256, 0, stream>>>(
        inp, NHID, p_wk, kx, DK, NHID, bk, 0, 0, 0);
    // vx = x@wv         (M=16384, K=1536, N=1024, bf16 out)
    k_gemm<false, true><<<dim3(ATT_OUT / 16, MROWS / 128, 1), 256, 0, stream>>>(
        inp, NHID, p_wv, vx, ATT_OUT, NHID, nullptr, 0, 0, 0);
    // GIX[g,k] = vx@wi[g,k]  (18 mats, M=16384, K=1024, N=256, fp32 out)
    k_gemm<true, false><<<dim3(BS / 16, MROWS / 128, 3 * NB), 256, 0, stream>>>(
        vx, ATT_OUT, p_wi, GIX, BS, ATT_OUT, nullptr,
        0, (long)ATT_OUT * BS, (long)MROWS * BS);

    // sequential recurrence: one fused kernel per timestep
    for (int t = 0; t < T_DIM; ++t) {
        k_step<<<4, 1024, 0, stream>>>(t, h_cur, h_bf, hx_ws, kx, gic,
                                       p_wh, p_wq, p_cq, p_ck, p_cv, p_cfc,
                                       bq, bk, bh, cbq, cbk, cbv, cbfc,
                                       ln_g, ln_b, GIX, out);
    }

    // h_final = last hidden state (second output in tuple)
    hipMemcpyAsync(out + (size_t)T_DIM * B_DIM * NHID, h_cur,
                   (size_t)B_DIM * NHID * sizeof(float),
                   hipMemcpyDeviceToDevice, stream);
}
